// GNN_autoformer_80951543595621
// MI455X (gfx1250) — matrix operand order repaired and checked by reading, not compiled, not run on hardware
//
#include <hip/hip_runtime.h>
#include <hip/hip_bf16.h>

// ---------------- problem constants ----------------
#define B_   32
#define T_   96
#define N_   1024
#define LBL  48
#define PRD  24
#define DEC2 512
#define HROWS (3 * B_ * PRD)   // 2304 stacked decoder rows

typedef _Float16 v16h __attribute__((ext_vector_type(16)));
typedef _Float16 v8h  __attribute__((ext_vector_type(8)));
typedef float    v8f  __attribute__((ext_vector_type(8)));

// ---------------- WMMA fragment helpers (gfx1250 wave32 layouts) ----------------
// A (16xK, row-major, element h: K = 16*(h/8) + 8*(lane>=16) + h%8)
__device__ __forceinline__ v16h load_a_frag(const _Float16* A, int lda, int lane) {
    int m = lane & 15, hi = lane >> 4;
    const _Float16* p = A + (size_t)m * lda + hi * 8;
    v8h lo = *(const v8h*)(p);
    v8h hh = *(const v8h*)(p + 16);
    v16h r;
#pragma unroll
    for (int i = 0; i < 8; ++i) { r[i] = lo[i]; r[8 + i] = hh[i]; }
    return r;
}
// B (Kx16) stored transposed Bt[col][k]; element h: K = 16*(h/8) + 8*(lane>=16) + h%8
__device__ __forceinline__ v16h load_b_frag(const _Float16* Bt, int ldb, int lane) {
    int n = lane & 15, hi = lane >> 4;
    const _Float16* p = Bt + (size_t)n * ldb + hi * 8;
    v8h lo = *(const v8h*)(p);
    v8h hh = *(const v8h*)(p + 16);
    v16h r;
#pragma unroll
    for (int i = 0; i < 8; ++i) { r[i] = lo[i]; r[8 + i] = hh[i]; }
    return r;
}
__device__ __forceinline__ v8f wmma16(v16h a, v16h b, v8f c) {
    return __builtin_amdgcn_wmma_f32_16x16x32_f16(false, a, false, b, (short)0, c, false, false);
}

// ---------------- small prep kernels ----------------
// u[b,n] = sum_t x[b,t,n]*W1[t] ; v[b,n] = W3 * sum_t x[b,t,n]*W2[t]
__global__ void prep_uv(const float* __restrict__ x, const float* __restrict__ W1,
                        const float* __restrict__ W2, const float* __restrict__ W3,
                        float* __restrict__ u, float* __restrict__ v) {
    int n = blockIdx.x * 256 + threadIdx.x;
    int b = blockIdx.y;
    float su = 0.f, sv = 0.f;
    const float* xb = x + (size_t)b * T_ * N_ + n;
    for (int t = 0; t < T_; ++t) {
        float xv = xb[(size_t)t * N_];
        su += xv * W1[t];
        sv += xv * W2[t];
    }
    u[b * N_ + n] = su;
    v[b * N_ + n] = W3[0] * sv;
}

__global__ void cvt_f32_f16(const float* __restrict__ src, _Float16* __restrict__ dst) {
    size_t i = (size_t)blockIdx.x * 256 + threadIdx.x;
    dst[i] = (_Float16)src[i];
}

// copy first LBL rows of y (B, LBL+PRD, N) -> yh (B, LBL, N) as f16
__global__ void cvt_y(const float* __restrict__ y, _Float16* __restrict__ yh) {
    size_t i = (size_t)blockIdx.x * 256 + threadIdx.x;  // over B*LBL*N
    int n = (int)(i & (N_ - 1));
    int l = (int)((i >> 10) % LBL);
    int b = (int)(i / ((size_t)LBL * N_));
    yh[i] = (_Float16)y[((size_t)b * (LBL + PRD) + l) * N_ + n];
}

// Wt[j*K+k] = (f16) W[k*Nw + j]   (transpose-convert decoder weights)
__global__ void twcvt(const float* __restrict__ W, _Float16* __restrict__ Wt, int K, int Nw) {
    size_t i = (size_t)blockIdx.x * 256 + threadIdx.x;  // over K*Nw (output order)
    int j = (int)(i / K);
    int k = (int)(i % K);
    Wt[i] = (_Float16)W[(size_t)k * Nw + j];
}

// ---------------- fused attention + neighbor aggregation ----------------
// One block = (b, 16-row block of S). Builds normalized S rows in LDS (f16),
// then WMMA's them against x^T (6 tiles) and y^T (3 tiles, only first 48 rows
// of y are ever used). Handles both adjacencies; exp only where adj != 0.
__global__ __launch_bounds__(128) void fused_att_agg(
    const float* __restrict__ u, const float* __restrict__ v,
    const float* __restrict__ bs,
    const float* __restrict__ adj1, const float* __restrict__ adj2,
    const _Float16* __restrict__ xh, const _Float16* __restrict__ yh,
    _Float16* __restrict__ x1h, _Float16* __restrict__ y1h,
    _Float16* __restrict__ x2h, _Float16* __restrict__ y2h)
{
    __shared__ _Float16 sS[16 * 1032];   // padded rows (bank-stagger), 33 KB
    __shared__ float    vL[N_];
    __shared__ float    uL[16];
    __shared__ float    rs[16];

    const int nblk = blockIdx.x;           // 0..63
    const int b    = blockIdx.y;           // 0..31
    const int tid  = threadIdx.x;
    const int lane = tid & 31, wave = tid >> 5;

    for (int i = tid; i < N_; i += 128) vL[i] = v[b * N_ + i];
    if (tid < 16) uL[tid] = u[b * N_ + nblk * 16 + tid];
    __syncthreads();

    for (int a = 0; a < 2; ++a) {
        const float* adj = a ? adj2 : adj1;
        _Float16* xo = a ? x2h : x1h;
        _Float16* yo = a ? y2h : y1h;

        // masked e = exp(sigmoid(u*v + bs)) * adj, with row sums (wave per row)
        for (int r = wave; r < 16; r += 4) {
            const int ng = nblk * 16 + r;
            const float un = uL[r];
            const float* arow  = adj + (size_t)ng * N_;
            const float* bsrow = bs  + (size_t)ng * N_;
            float sum = 0.f;
            for (int m = lane; m < N_; m += 32) {
                float val = 0.f;
                if (arow[m] != 0.0f) {
                    float p  = un * vL[m] + bsrow[m];
                    float sg = 1.0f / (1.0f + __expf(-p));
                    val = __expf(sg);
                }
                sS[r * 1032 + m] = (_Float16)val;
                sum += val;
            }
#pragma unroll
            for (int off = 16; off; off >>= 1) sum += __shfl_xor(sum, off, 32);
            if (lane == 0) rs[r] = 1.0f / sum;   // diagonal guarantees sum >= 1
        }
        __syncthreads();
        // normalize rows in place
        for (int i = tid; i < 16 * N_; i += 128) {
            int n = i >> 10, m = i & (N_ - 1);
            sS[n * 1032 + m] = (_Float16)((float)sS[n * 1032 + m] * rs[n]);
        }
        __syncthreads();

        // 9 output tiles: 6 for x (T=96), 3 for y (first 48 rows)
        for (int tile = wave; tile < 9; tile += 4) {
            const bool isY = tile >= 6;
            const int  t0  = (isY ? (tile - 6) : tile) * 16;
            const _Float16* Bt = isY ? (yh + ((size_t)b * LBL + t0) * N_)
                                     : (xh + ((size_t)b * T_  + t0) * N_);
            v8f acc = {};
            for (int k0 = 0; k0 < N_; k0 += 32) {
                v16h af = load_a_frag(&sS[k0], 1032, lane);  // A = S rows (M=n, K=m)
                v16h bf = load_b_frag(Bt + k0, N_, lane);    // B[k=m][col=t] = X[t,m]
                acc = wmma16(af, bf, acc);
            }
            const int t  = t0 + (lane & 15);
            const int hi = lane >> 4;
            _Float16* dst = (isY ? (yo + ((size_t)b * LBL + t) * N_)
                                 : (xo + ((size_t)b * T_  + t) * N_))
                            + nblk * 16 + hi * 8;
            v8h o;
#pragma unroll
            for (int r = 0; r < 8; ++r) o[r] = (_Float16)acc[r];
            *(v8h*)dst = o;
        }
        __syncthreads();
    }
}

// ---------------- stubbed Autoformer time-mixer ----------------
// H[b,p,n] = sum_t ew[p,t]*X[b,t,n] + sum_{l<48} dw[p,l]*Y[b,l,n]  (f16 out)
template <typename TIn>
__global__ void autoformer_k(const TIn* __restrict__ X, const TIn* __restrict__ Y,
                             const float* __restrict__ ew, const float* __restrict__ dw,
                             _Float16* __restrict__ H, int strideXb, int strideYb) {
    int n = blockIdx.x * 256 + threadIdx.x;
    int p = blockIdx.y;
    int b = blockIdx.z;
    float acc = 0.f;
    const TIn* xb = X + (size_t)b * strideXb + n;
    for (int t = 0; t < T_; ++t) acc += ew[p * T_ + t] * (float)xb[(size_t)t * N_];
    const TIn* yb = Y + (size_t)b * strideYb + n;
    for (int l = 0; l < LBL; ++l) acc += dw[p * (LBL + PRD) + l] * (float)yb[(size_t)l * N_];
    H[((size_t)b * PRD + p) * N_ + n] = (_Float16)acc;
}

// ---------------- generic WMMA GEMM: C = act(A @ Bt^T + bias) ----------------
// A: MxK f16 row-major; Bt: NxK f16 (weights pre-transposed). Wave = 16x64 strip.
// MODE 0: f32 out (+bias). MODE 1: f16 out, relu(+bias).
template <int MODE>
__global__ void __launch_bounds__(32) gemm_wmma(
    const _Float16* __restrict__ A, const _Float16* __restrict__ Bt,
    const float* __restrict__ bias, void* __restrict__ Cout,
    int Nn, int K)
{
    const int lane = threadIdx.x;
    const int row0 = blockIdx.y * 16;
    const int col0 = blockIdx.x * 64;
    const _Float16* Ap = A  + (size_t)row0 * K;
    const _Float16* Bp = Bt + (size_t)col0 * K;
    v8f z = {};
    v8f acc[4] = {z, z, z, z};
    for (int k0 = 0; k0 < K; k0 += 32) {
        v16h a = load_a_frag(Ap + k0, K, lane);
#pragma unroll
        for (int j = 0; j < 4; ++j) {
            v16h bf = load_b_frag(Bp + (size_t)(16 * j) * K + k0, K, lane);
            acc[j] = wmma16(a, bf, acc[j]);
        }
    }
    const int nc = lane & 15, hi = lane >> 4;
#pragma unroll
    for (int j = 0; j < 4; ++j) {
        const int col = col0 + 16 * j + nc;
        const float bv = bias[col];
#pragma unroll
        for (int r = 0; r < 8; ++r) {
            const int row = row0 + r + 8 * hi;
            float val = acc[j][r] + bv;
            if (MODE == 1) {
                ((_Float16*)Cout)[(size_t)row * Nn + col] = (_Float16)fmaxf(val, 0.0f);
            } else {
                ((float*)Cout)[(size_t)row * Nn + col] = val;
            }
        }
    }
}

// ---------------- final coefficient combine ----------------
__global__ void combine_k(const float* __restrict__ G3, const float* __restrict__ coeff,
                          float* __restrict__ out) {
    size_t i = (size_t)blockIdx.x * 256 + threadIdx.x;   // over 768*1024
    const size_t stride = (size_t)B_ * PRD * N_;
    out[i] = coeff[0] * G3[i] + coeff[1] * G3[i + stride] + coeff[2] * G3[i + 2 * stride];
}

// ---------------- host launcher ----------------
extern "C" void kernel_launch(void* const* d_in, const int* in_sizes, int n_in,
                              void* d_out, int out_size, void* d_ws, size_t ws_size,
                              hipStream_t stream) {
    (void)in_sizes; (void)n_in; (void)out_size; (void)ws_size;
    const float* x      = (const float*)d_in[0];
    const float* y      = (const float*)d_in[1];
    const float* adj1   = (const float*)d_in[2];
    const float* adj2   = (const float*)d_in[3];
    const float* W1     = (const float*)d_in[4];
    const float* W2     = (const float*)d_in[5];
    const float* W3     = (const float*)d_in[6];
    const float* bs     = (const float*)d_in[7];
    const float* enc_w1 = (const float*)d_in[8];
    const float* dec_w1 = (const float*)d_in[9];
    const float* enc_w2 = (const float*)d_in[10];
    const float* dec_w2 = (const float*)d_in[11];
    const float* enc_w3 = (const float*)d_in[12];
    const float* dec_w3 = (const float*)d_in[13];
    const float* coeff  = (const float*)d_in[14];
    const float* d1_w   = (const float*)d_in[15];
    const float* d1_b   = (const float*)d_in[16];
    const float* d2_w   = (const float*)d_in[17];
    const float* d2_b   = (const float*)d_in[18];
    const float* d3_w   = (const float*)d_in[19];
    const float* d3_b   = (const float*)d_in[20];
    float* out = (float*)d_out;

    // workspace carve-up (~54 MB, fits comfortably in the 192 MB L2)
    size_t off = 0;
    auto alloc = [&](size_t bytes) -> void* {
        void* p = (char*)d_ws + off;
        off += (bytes + 255) & ~(size_t)255;
        return p;
    };
    float*     u    = (float*)alloc((size_t)B_ * N_ * 4);
    float*     v    = (float*)alloc((size_t)B_ * N_ * 4);
    _Float16*  xh   = (_Float16*)alloc((size_t)B_ * T_  * N_ * 2);
    _Float16*  yh   = (_Float16*)alloc((size_t)B_ * LBL * N_ * 2);
    _Float16*  x1h  = (_Float16*)alloc((size_t)B_ * T_  * N_ * 2);
    _Float16*  x2h  = (_Float16*)alloc((size_t)B_ * T_  * N_ * 2);
    _Float16*  y1h  = (_Float16*)alloc((size_t)B_ * LBL * N_ * 2);
    _Float16*  y2h  = (_Float16*)alloc((size_t)B_ * LBL * N_ * 2);
    _Float16*  Wt1  = (_Float16*)alloc((size_t)1024 * 1024 * 2);  // [j<1024][k<1024]
    _Float16*  Wt2  = (_Float16*)alloc((size_t)512  * 1024 * 2);  // [j<512 ][k<1024]
    _Float16*  Wt3  = (_Float16*)alloc((size_t)1024 * 512  * 2);  // [j<1024][k<512 ]
    _Float16*  H    = (_Float16*)alloc((size_t)HROWS * N_ * 2);
    _Float16*  G1   = (_Float16*)alloc((size_t)HROWS * N_ * 2);
    _Float16*  G2   = (_Float16*)alloc((size_t)HROWS * DEC2 * 2);
    float*     G3   = (float*)alloc((size_t)HROWS * N_ * 4);

    // 1) rank-1 attention vectors + f16 conversions + weight transposes
    prep_uv<<<dim3(N_ / 256, B_), 256, 0, stream>>>(x, W1, W2, W3, u, v);
    cvt_f32_f16<<<(B_ * T_ * N_) / 256, 256, 0, stream>>>(x, xh);
    cvt_y<<<(B_ * LBL * N_) / 256, 256, 0, stream>>>(y, yh);
    twcvt<<<(1024 * 1024) / 256, 256, 0, stream>>>(d1_w, Wt1, 1024, 1024);
    twcvt<<<(1024 * 512 ) / 256, 256, 0, stream>>>(d2_w, Wt2, 1024, 512);
    twcvt<<<(512  * 1024) / 256, 256, 0, stream>>>(d3_w, Wt3, 512, 1024);

    // 2) fused masked-softmax + S@X^T / S@Y^T via WMMA (S never hits HBM)
    fused_att_agg<<<dim3(N_ / 16, B_), 128, 0, stream>>>(
        u, v, bs, adj1, adj2, xh, yh, x1h, y1h, x2h, y2h);

    // 3) time-mixers -> stacked decoder input H (2304 x 1024, f16)
    autoformer_k<float><<<dim3(N_ / 256, PRD, B_), 256, 0, stream>>>(
        x, y, enc_w1, dec_w1, H, T_ * N_, (LBL + PRD) * N_);
    autoformer_k<_Float16><<<dim3(N_ / 256, PRD, B_), 256, 0, stream>>>(
        x1h, y1h, enc_w2, dec_w2, H + (size_t)B_ * PRD * N_, T_ * N_, LBL * N_);
    autoformer_k<_Float16><<<dim3(N_ / 256, PRD, B_), 256, 0, stream>>>(
        x2h, y2h, enc_w3, dec_w3, H + (size_t)2 * B_ * PRD * N_, T_ * N_, LBL * N_);

    // 4) shared decoder MLP, batched over all three ensemble members
    gemm_wmma<1><<<dim3(1024 / 64, HROWS / 16), 32, 0, stream>>>(H,  Wt1, d1_b, G1, 1024, 1024);
    gemm_wmma<1><<<dim3(512  / 64, HROWS / 16), 32, 0, stream>>>(G1, Wt2, d2_b, G2, 512, 1024);
    gemm_wmma<0><<<dim3(1024 / 64, HROWS / 16), 32, 0, stream>>>(G2, Wt3, d3_b, G3, 1024, 512);

    // 5) coeff-weighted ensemble combine
    combine_k<<<(B_ * PRD * N_) / 256, 256, 0, stream>>>(G3, coeff, out);
}
